// PixPro_43705587204329
// MI455X (gfx1250) — matrix operand order repaired and checked by reading, not compiled, not run on hardware
//
#include <hip/hip_runtime.h>
#include <stdint.h>
#include <math.h>

#define BATCH   2048
#define CCH     256
#define HW      49
#define PADP    64
#define KC      64          // K-chunk (channels per LDS stage)
#define NCHUNK  (CCH/KC)
#define NTHREADS 128
#define FRWORDS (2*2*64*8)  // [ks][half][col][8 words] per array per chunk

typedef __attribute__((ext_vector_type(16))) __bf16    v16bf;
typedef __attribute__((ext_vector_type(8)))  float     v8f;
typedef __attribute__((ext_vector_type(8)))  uint32_t  u32x8;

union frag_u { u32x8 w; uint32_t u[8]; v16bf v; };

__device__ __forceinline__ uint16_t f2bf_rn(float f) {
    uint32_t u = __float_as_uint(f);
    u += 0x7FFFu + ((u >> 16) & 1u);      // round-to-nearest-even
    return (uint16_t)(u >> 16);
}
__device__ __forceinline__ float bf2f(uint32_t half16) {
    return __uint_as_float(half16 << 16);
}

__global__ __launch_bounds__(NTHREADS)
void pixpro_cosloss_kernel(const float* __restrict__ f1g,
                           const float* __restrict__ f2g,
                           const int*   __restrict__ baseA,
                           const int*   __restrict__ momA,
                           float*       __restrict__ out)
{
    // Fragment-major bf16 hi/lo tiles: f1 in WMMA-A order, f2 in WMMA-B order.
    __shared__ uint32_t lAh[FRWORDS];
    __shared__ uint32_t lAl[FRWORDS];
    __shared__ uint32_t lBh[FRWORDS];
    __shared__ uint32_t lBl[FRWORDS];
    __shared__ float n1s[PADP];
    __shared__ float n2s[PADP];
    __shared__ float red[NTHREADS];

    const int tid  = threadIdx.x;
    const int lane = tid & 31;
    const int wave = tid >> 5;
    const int h    = lane >> 4;           // lane half (0/1)
    const int lm   = lane & 15;
    const int b    = blockIdx.x;

    const float* f1 = f1g + (size_t)b * (CCH * HW);
    const float* f2 = f2g + (size_t)b * (CCH * HW);

    const int mcol = (wave << 4) + lm;    // this wave's A (M) column

    float nsum = 0.f;                     // tid<49: n1 partial; 64<=tid<113: n2 partial
    v8f acc[4];
    #pragma unroll
    for (int nt = 0; nt < 4; ++nt)
        #pragma unroll
        for (int r = 0; r < 8; ++r) acc[nt][r] = 0.f;

    uint16_t* sAh = reinterpret_cast<uint16_t*>(lAh);
    uint16_t* sAl = reinterpret_cast<uint16_t*>(lAl);
    uint16_t* sBh = reinterpret_cast<uint16_t*>(lBh);
    uint16_t* sBl = reinterpret_cast<uint16_t*>(lBl);

    for (int kc = 0; kc < NCHUNK; ++kc) {
        const int k0 = kc * KC;

        // ---- stage one K-chunk: convert f32 -> bf16 hi/lo, scatter in fragment order
        for (int e = tid; e < KC * PADP; e += NTHREADS) {
            const int k = e >> 6;         // 0..63 (channel within chunk)
            const int m = e & 63;         // padded pixel column
            const float x1 = (m < HW) ? f1[(k0 + k) * HW + m] : 0.f;
            const float x2 = (m < HW) ? f2[(k0 + k) * HW + m] : 0.f;
            const uint16_t h1 = f2bf_rn(x1);
            const uint16_t o1 = f2bf_rn(x1 - bf2f(h1));
            const uint16_t h2 = f2bf_rn(x2);
            const uint16_t o2 = f2bf_rn(x2 - bf2f(h2));

            const int kp   = k >> 1;      // packed K-pair index (0..31)
            const int hsel = k & 1;       // low/high half of the dword
            const int ks   = kp >> 4;     // 32-wide K step (0/1)
            const int r    = kp & 15;
            // A layout: kp = 16ks + 8*h + v
            const int va = r & 7;
            const int ha = r >> 3;
            const int ai = (((((ks << 1) | ha) << 6) | m) << 3) | va;
            // B layout: kp = 16ks + 8*h + v
            const int vb = r & 7;
            const int hb = r >> 3;
            const int bi = (((((ks << 1) | hb) << 6) | m) << 3) | vb;

            sAh[(ai << 1) | hsel] = h1;  sAl[(ai << 1) | hsel] = o1;
            sBh[(bi << 1) | hsel] = h2;  sBl[(bi << 1) | hsel] = o2;
        }

        // prefetch next chunk while this one is consumed (global_prefetch_b8)
        if (kc + 1 < NCHUNK) {
            const float* nf1 = f1 + (k0 + KC) * HW;
            const float* nf2 = f2 + (k0 + KC) * HW;
            for (int off = tid * 32; off < KC * HW; off += NTHREADS * 32) {
                __builtin_prefetch(nf1 + off, 0, 0);
                __builtin_prefetch(nf2 + off, 0, 0);
            }
        }
        __syncthreads();

        // ---- norm partials (sum of squares over all K of the chunk; order-free)
        if (tid < HW) {
            #pragma unroll
            for (int blk = 0; blk < 4; ++blk) {
                frag_u wh, wl;
                wh.w = *(const u32x8*)&lAh[((blk << 6) | tid) << 3];
                wl.w = *(const u32x8*)&lAl[((blk << 6) | tid) << 3];
                #pragma unroll
                for (int v = 0; v < 8; ++v) {
                    const float x0 = bf2f(wh.u[v] & 0xFFFFu) + bf2f(wl.u[v] & 0xFFFFu);
                    const float x1 = bf2f(wh.u[v] >> 16)     + bf2f(wl.u[v] >> 16);
                    nsum += x0 * x0 + x1 * x1;
                }
            }
        } else if (tid >= 64 && tid < 64 + HW) {
            const int m = tid - 64;
            #pragma unroll
            for (int blk = 0; blk < 4; ++blk) {
                frag_u wh, wl;
                wh.w = *(const u32x8*)&lBh[((blk << 6) | m) << 3];
                wl.w = *(const u32x8*)&lBl[((blk << 6) | m) << 3];
                #pragma unroll
                for (int v = 0; v < 8; ++v) {
                    const float x0 = bf2f(wh.u[v] & 0xFFFFu) + bf2f(wl.u[v] & 0xFFFFu);
                    const float x1 = bf2f(wh.u[v] >> 16)     + bf2f(wl.u[v] >> 16);
                    nsum += x0 * x0 + x1 * x1;
                }
            }
        }

        // ---- WMMA: contiguous fragment loads (ds_load_b128 pairs) + bf16x3 products
        #pragma unroll
        for (int ks = 0; ks < KC / 32; ++ks) {
            frag_u ah, al;
            const int abase = (((((ks << 1) | h) << 6) | mcol) << 3);
            ah.w = *(const u32x8*)&lAh[abase];
            al.w = *(const u32x8*)&lAl[abase];
            #pragma unroll
            for (int nt = 0; nt < 4; ++nt) {
                const int ncol  = (nt << 4) + lm;
                const int bbase = (((((ks << 1) | h) << 6) | ncol) << 3);
                frag_u bh, bl;
                bh.w = *(const u32x8*)&lBh[bbase];
                bl.w = *(const u32x8*)&lBl[bbase];
                // (ah+al)(bh+bl) ~= ah*bh + ah*bl + al*bh  (fp32-like precision)
                acc[nt] = __builtin_amdgcn_wmma_f32_16x16x32_bf16(
                    false, ah.v, false, bh.v, (short)0, acc[nt], false, false);
                acc[nt] = __builtin_amdgcn_wmma_f32_16x16x32_bf16(
                    false, ah.v, false, bl.v, (short)0, acc[nt], false, false);
                acc[nt] = __builtin_amdgcn_wmma_f32_16x16x32_bf16(
                    false, al.v, false, bh.v, (short)0, acc[nt], false, false);
            }
        }
        __syncthreads();
    }

    // ---- finalize norms ----
    if (tid < HW)                      n1s[tid]      = sqrtf(nsum);
    if (tid >= 64 && tid < 64 + HW)    n2s[tid - 64] = sqrtf(nsum);
    __syncthreads();

    // ---- epilogue: cos = G/max(n1*n2,eps); weight = baseA[m,n] + momA[n,m] ----
    float local = 0.f;
    #pragma unroll
    for (int nt = 0; nt < 4; ++nt) {
        const int n = (nt << 4) + lm;
        if (n < HW) {
            #pragma unroll
            for (int r = 0; r < 8; ++r) {
                const int m = (wave << 4) + h * 8 + r;   // C/D layout: M = 8*half + r
                if (m < HW) {
                    const float denom = fmaxf(n1s[m] * n2s[n], 1e-6f);
                    const float wgt = (baseA[m * HW + n] != 0 ? 1.f : 0.f)
                                    + (momA[n * HW + m] != 0 ? 1.f : 0.f);
                    local += (acc[nt][r] / denom) * wgt;
                }
            }
        }
    }

    red[tid] = local;
    __syncthreads();
    #pragma unroll
    for (int s = NTHREADS / 2; s > 0; s >>= 1) {
        if (tid < s) red[tid] += red[tid + s];
        __syncthreads();
    }
    if (tid == 0) out[b] = -red[0] * (1.0f / 2401.0f);
}

extern "C" void kernel_launch(void* const* d_in, const int* in_sizes, int n_in,
                              void* d_out, int out_size, void* d_ws, size_t ws_size,
                              hipStream_t stream) {
    const float* base_feature   = (const float*)d_in[0];
    const float* moment_feature = (const float*)d_in[1];
    const int*   base_A         = (const int*)d_in[2];
    const int*   moment_A       = (const int*)d_in[3];
    float* out = (float*)d_out;

    dim3 grid(BATCH), block(NTHREADS);
    hipLaunchKernelGGL(pixpro_cosloss_kernel, grid, block, 0, stream,
                       base_feature, moment_feature, base_A, moment_A, out);
}
